// Transformer_3642132267242
// MI455X (gfx1250) — matrix operand order repaired and checked by reading, not compiled, not run on hardware
//
#include <hip/hip_runtime.h>

typedef __attribute__((ext_vector_type(16))) __bf16 v16bf;
typedef __attribute__((ext_vector_type(8)))  __bf16 v8bf;
typedef __attribute__((ext_vector_type(8)))  float  v8f;

constexpr int Bsz = 4, T = 1024, D = 1024, H = 16, HS = 64, Lyr = 4, V = 32000, DFF = 4096;
constexpr int MR = Bsz * T;   // 4096 token rows

static __device__ __forceinline__ __bf16 f2bf(float f) {
    union { float f; unsigned u; } x; x.f = f;
    unsigned r = (x.u + 0x7FFFu + ((x.u >> 16) & 1u)) >> 16;
    union { unsigned short s; __bf16 b; } y; y.s = (unsigned short)r;
    return y.b;
}

// low 32 bits of a generic pointer into the LDS aperture == DS byte address
static __device__ __forceinline__ unsigned lds_off(const void* p) {
    return (unsigned)(unsigned long long)p;
}

// per-lane 16-byte async copy global -> LDS (ASYNCcnt-tracked DMA path)
static __device__ __forceinline__ void async_copy16(unsigned lds_addr, const void* g) {
    asm volatile("global_load_async_to_lds_b128 %0, %1, off"
                 :: "v"(lds_addr), "v"((unsigned long long)g)
                 : "memory");
}
static __device__ __forceinline__ void wait_async0() {
    asm volatile("s_wait_asynccnt 0x0" ::: "memory");
}

// ---------------- embedding: h = tok_emb[x] + pos_emb ----------------
__global__ void k_embed(const int* __restrict__ x, const float* __restrict__ tok,
                        const float* __restrict__ pos, float* __restrict__ h) {
    size_t i = (size_t)blockIdx.x * blockDim.x + threadIdx.x;
    if (i >= (size_t)MR * D) return;
    int d = (int)(i % D);
    int row = (int)(i / D);
    int t = row % T;
    h[i] = tok[(size_t)x[row] * D + d] + pos[(size_t)t * D + d];
}

// ------------- weight convert+transpose: out[b][n][k] = in[b][k][n] (fp32->bf16) -------------
__global__ void k_transpose_bf16(const float* __restrict__ in, __bf16* __restrict__ out,
                                 int K, int N, size_t total) {
    size_t i = (size_t)blockIdx.x * blockDim.x + threadIdx.x;
    if (i >= total) return;
    int k = (int)(i % K);
    size_t t = i / K;
    int n = (int)(t % N);
    size_t b = t / N;
    out[i] = f2bf(in[(b * (size_t)K + k) * N + n]);
}

// ---------------- layernorm (fp32 in, bf16 out), one block per row ----------------
__global__ __launch_bounds__(256) void k_layernorm_bf16(const float* __restrict__ x,
                                                        const float* __restrict__ g,
                                                        const float* __restrict__ b,
                                                        __bf16* __restrict__ out) {
    __shared__ float red[256];
    const int row = blockIdx.x, tid = threadIdx.x;
    const float* xr = x + (size_t)row * D;
    float s = 0.f;
    for (int i = tid; i < D; i += 256) s += xr[i];
    red[tid] = s; __syncthreads();
    for (int st = 128; st > 0; st >>= 1) { if (tid < st) red[tid] += red[tid + st]; __syncthreads(); }
    float mean = red[0] / (float)D;
    __syncthreads();
    float v = 0.f;
    for (int i = tid; i < D; i += 256) { float d = xr[i] - mean; v += d * d; }
    red[tid] = v; __syncthreads();
    for (int st = 128; st > 0; st >>= 1) { if (tid < st) red[tid] += red[tid + st]; __syncthreads(); }
    float inv = rsqrtf(red[0] / (float)D + 1e-5f);
    for (int i = tid; i < D; i += 256)
        out[(size_t)row * D + i] = f2bf((xr[i] - mean) * inv * g[i] + b[i]);
}

// ---------------- bf16 WMMA GEMM: C[M][N] = A[M][K] * Bt[N][K]^T (+bias,+res,relu) ----------------
// block = 256 threads = 8 waves; wave tile = 16(M) x 64(N); K step 32.
// B tile (64 cols x 32 k) staged in LDS via async global->LDS DMA, double buffered:
// each 16B chunk of B is fetched from L2 once per block instead of once per wave (8x saving).
__global__ __launch_bounds__(256) void k_gemm_bf16(
    const __bf16* __restrict__ A, const __bf16* __restrict__ Bt,
    const float* __restrict__ bias, const float* __restrict__ res,
    float* __restrict__ Cf, __bf16* __restrict__ Cb,
    int M, int N, int K, int relu) {
    __shared__ __align__(16) __bf16 Bs[2][64 * 32];   // [buf][col][k], 4 KB each
    const int tid  = threadIdx.x;
    const int lane = tid & 31;
    const int wave = tid >> 5;
    const int hf   = lane >> 4;     // which 16-lane half
    const int l16  = lane & 15;
    const int mbase = blockIdx.y * 128 + wave * 16;
    const int nbase = blockIdx.x * 64;
    (void)M;

    // async B-tile loader: thread t covers col = t/4, k-part = t%4 (8 bf16 = 16 B)
    const int bcol  = tid >> 2;
    const int kpart = tid & 3;
    const __bf16* bsrc = Bt + (size_t)(nbase + bcol) * K + kpart * 8;
    const unsigned bdst0 = lds_off(&Bs[0][bcol * 32 + kpart * 8]);
    const unsigned bdst1 = lds_off(&Bs[1][bcol * 32 + kpart * 8]);

    async_copy16(bdst0, bsrc);            // stage first K tile
    wait_async0();
    __syncthreads();

    v8f acc[4] = {};
    const __bf16* arow = A + (size_t)(mbase + l16) * K;
    int buf = 0;
    for (int kk = 0; kk < K; kk += 32) {
        if (kk + 32 < K) {                                   // uniform branch: EXEC stays full
            async_copy16(buf ? bdst0 : bdst1, bsrc + kk + 32);
            __builtin_prefetch(arow + kk + 64, 0, 1);        // next A chunk -> global_prefetch
        }
        union { v16bf v; v8bf h[2]; } a;
        a.h[0] = *(const v8bf*)(arow + kk + hf * 16);        // K = kk + hf*16 .. +7
        a.h[1] = *(const v8bf*)(arow + kk + hf * 16 + 8);    // K = kk+hf*16+8 .. +7
        const __bf16* bb = &Bs[buf][0];
#pragma unroll
        for (int n = 0; n < 4; n++) {
            v16bf bfrag = *(const v16bf*)(bb + (n * 16 + l16) * 32 + hf * 16);
            acc[n] = __builtin_amdgcn_wmma_f32_16x16x32_bf16(
                false, a.v, false, bfrag, (short)0, acc[n], false, false);
        }
        wait_async0();          // own wave's in-flight tile complete
        __syncthreads();        // all waves' tile parts visible + done reading old buf
        buf ^= 1;
    }
#pragma unroll
    for (int n = 0; n < 4; n++) {
        const int col = nbase + n * 16 + l16;
        const float bv = bias ? bias[col] : 0.f;
#pragma unroll
        for (int r = 0; r < 8; r++) {
            const int row = mbase + r + 8 * hf;
            float v = acc[n][r] + bv;
            if (res)  v += res[(size_t)row * N + col];
            if (relu) v = fmaxf(v, 0.f);
            if (Cf) Cf[(size_t)row * N + col] = v;
            if (Cb) Cb[(size_t)row * N + col] = f2bf(v);
        }
    }
}

// ---------------- fused causal attention for one (b,h,16-row q block); 1 wave ----------------
__global__ __launch_bounds__(32) void k_attention(
    const __bf16* __restrict__ q, const __bf16* __restrict__ k,
    const __bf16* __restrict__ v, __bf16* __restrict__ att) {
    __shared__ float S[16 * 1024];           // 64 KB score row-block
    const int lane = threadIdx.x & 31;
    const int hf = lane >> 4;
    const int l16 = lane & 15;
    const int qb = blockIdx.x;               // 0..T/16-1
    const int b  = blockIdx.y >> 4;
    const int hh = blockIdx.y & 15;
    const float scale = 0.125f;              // HS^-0.5
    const int LD = H * HS;                   // 1024
    const size_t base = (size_t)b * T * LD + hh * HS;
    const __bf16* Qp = q + base;
    const __bf16* Kp = k + base;
    const __bf16* Vp = v + base;
    const int iend = qb * 16 + 15;
    const int jmax = ((iend + 1 + 31) >> 5) << 5;   // causal bound rounded to 32

    // Q A-fragments (row i = qb*16 + l16), two K-chunks of 32 over HS=64
    union { v16bf v; v8bf h[2]; } qa[2];
    const __bf16* qrow = Qp + (size_t)(qb * 16 + l16) * LD;
#pragma unroll
    for (int c = 0; c < 2; c++) {
        qa[c].h[0] = *(const v8bf*)(qrow + c * 32 + hf * 16);
        qa[c].h[1] = *(const v8bf*)(qrow + c * 32 + hf * 16 + 8);
    }

    // scores: S[m][j] = scale * q_i . k_j  (masked)
    for (int jt = 0; jt * 16 < jmax; jt++) {
        v8f s = {};
#pragma unroll
        for (int c = 0; c < 2; c++) {
            const __bf16* kp = Kp + (size_t)(jt * 16 + l16) * LD + c * 32 + hf * 16;
            v16bf bfrag = *(const v16bf*)kp;
            s = __builtin_amdgcn_wmma_f32_16x16x32_bf16(
                false, qa[c].v, false, bfrag, (short)0, s, false, false);
        }
        const int j = jt * 16 + l16;
#pragma unroll
        for (int r = 0; r < 8; r++) {
            const int m = r + 8 * hf;
            const int i = qb * 16 + m;
            S[m * 1024 + j] = (j <= i) ? s[r] * scale : -1e30f;
        }
    }

    // softmax per row over [0, jmax)
    for (int m = 0; m < 16; m++) {
        float* row = S + m * 1024;
        float mx = -1e30f;
        for (int j = lane; j < jmax; j += 32) mx = fmaxf(mx, row[j]);
#pragma unroll
        for (int o = 16; o > 0; o >>= 1) mx = fmaxf(mx, __shfl_xor(mx, o, 32));
        float sum = 0.f;
        for (int j = lane; j < jmax; j += 32) sum += __expf(row[j] - mx);
#pragma unroll
        for (int o = 16; o > 0; o >>= 1) sum += __shfl_xor(sum, o, 32);
        const float inv = 1.f / sum;
        for (int j = lane; j < jmax; j += 32) row[j] = __expf(row[j] - mx) * inv;
    }

    // O = P @ V  (K-dim = keys, 32 per WMMA; N = HS in 4 tiles of 16)
    v8f o[4] = {};
    for (int kb = 0; kb * 32 < jmax; kb++) {
        v16bf pa;
#pragma unroll
        for (int e = 0; e < 8; e++) pa[e] = f2bf(S[l16 * 1024 + kb * 32 + hf * 16 + e]);
#pragma unroll
        for (int e = 0; e < 8; e++) pa[8 + e] = f2bf(S[l16 * 1024 + kb * 32 + hf * 16 + 8 + e]);
#pragma unroll
        for (int n = 0; n < 4; n++) {
            v16bf vb;
#pragma unroll
            for (int e = 0; e < 16; e++) {
                const int kk = kb * 32 + hf * 16 + e;
                vb[e] = Vp[(size_t)kk * LD + n * 16 + l16];
            }
            o[n] = __builtin_amdgcn_wmma_f32_16x16x32_bf16(
                false, pa, false, vb, (short)0, o[n], false, false);
        }
    }
#pragma unroll
    for (int n = 0; n < 4; n++) {
        const int col = hh * HS + n * 16 + l16;
#pragma unroll
        for (int r = 0; r < 8; r++) {
            const int t = qb * 16 + r + 8 * hf;
            att[((size_t)b * T + t) * LD + col] = f2bf(o[n][r]);
        }
    }
}

// ---------------- cross-entropy loss over logits rows ----------------
__global__ __launch_bounds__(256) void k_loss(const float* __restrict__ logits,
                                              const int* __restrict__ tgt,
                                              float* __restrict__ loss) {
    __shared__ float red[256];
    const int row = blockIdx.x, tid = threadIdx.x;
    const float* lr = logits + (size_t)row * V;
    float mx = -1e30f;
    for (int j = tid; j < V; j += 256) mx = fmaxf(mx, lr[j]);
    red[tid] = mx; __syncthreads();
    for (int st = 128; st > 0; st >>= 1) { if (tid < st) red[tid] = fmaxf(red[tid], red[tid + st]); __syncthreads(); }
    mx = red[0]; __syncthreads();
    float s = 0.f;
    for (int j = tid; j < V; j += 256) s += __expf(lr[j] - mx);
    red[tid] = s; __syncthreads();
    for (int st = 128; st > 0; st >>= 1) { if (tid < st) red[tid] += red[tid + st]; __syncthreads(); }
    if (tid == 0) {
        float term = mx + logf(red[0]) - lr[tgt[row]];
        atomicAdd(loss, term / (float)MR);
    }
}

extern "C" void kernel_launch(void* const* d_in, const int* in_sizes, int n_in,
                              void* d_out, int out_size, void* d_ws, size_t ws_size,
                              hipStream_t stream) {
    (void)in_sizes; (void)n_in; (void)out_size; (void)ws_size;
    const int*   x   = (const int*)d_in[0];
    const int*   tgt = (const int*)d_in[1];
    const float* tok = (const float*)d_in[2];
    const float* pos = (const float*)d_in[3];
    const float* qw  = (const float*)d_in[4];
    const float* kw  = (const float*)d_in[5];
    const float* vw  = (const float*)d_in[6];
    const float* pw  = (const float*)d_in[7];
    const float* pb  = (const float*)d_in[8];
    const float* g1  = (const float*)d_in[9];
    const float* b1  = (const float*)d_in[10];
    const float* g2  = (const float*)d_in[11];
    const float* b2  = (const float*)d_in[12];
    const float* w1  = (const float*)d_in[13];
    const float* fb1 = (const float*)d_in[14];
    const float* w2  = (const float*)d_in[15];
    const float* fb2 = (const float*)d_in[16];
    const float* gf  = (const float*)d_in[17];
    const float* bf  = (const float*)d_in[18];
    const float* ow  = (const float*)d_in[19];
    const float* ob  = (const float*)d_in[20];

    char* p = (char*)d_ws;
    auto carve = [&](size_t bytes) -> char* {
        char* r = p;
        p += (bytes + 255) & ~(size_t)255;
        return r;
    };
    float*  h    = (float*) carve((size_t)MR * D * 4);
    __bf16* hn   = (__bf16*)carve((size_t)MR * D * 2);
    __bf16* qbuf = (__bf16*)carve((size_t)MR * D * 2);
    __bf16* kbuf = (__bf16*)carve((size_t)MR * D * 2);
    __bf16* vbuf = (__bf16*)carve((size_t)MR * D * 2);
    __bf16* attb = (__bf16*)carve((size_t)MR * D * 2);
    __bf16* f1b  = (__bf16*)carve((size_t)MR * DFF * 2);
    __bf16* qt   = (__bf16*)carve((size_t)Lyr * D * D * 2);     // [l][h*HS][D]
    __bf16* kt   = (__bf16*)carve((size_t)Lyr * D * D * 2);
    __bf16* vt   = (__bf16*)carve((size_t)Lyr * D * D * 2);
    __bf16* pt   = (__bf16*)carve((size_t)Lyr * D * D * 2);
    __bf16* w1t  = (__bf16*)carve((size_t)Lyr * D * DFF * 2);
    __bf16* w2t  = (__bf16*)carve((size_t)Lyr * DFF * D * 2);
    __bf16* owt  = (__bf16*)carve((size_t)D * V * 2);

    auto tr = [&](const float* in, __bf16* out, int K, int N, int batch) {
        size_t total = (size_t)batch * K * N;
        int blocks = (int)((total + 255) / 256);
        k_transpose_bf16<<<blocks, 256, 0, stream>>>(in, out, K, N, total);
    };
    tr(qw, qt, D, HS, Lyr * H);
    tr(kw, kt, D, HS, Lyr * H);
    tr(vw, vt, D, HS, Lyr * H);
    tr(pw, pt, D, D, Lyr);
    tr(w1, w1t, D, DFF, Lyr);
    tr(w2, w2t, DFF, D, Lyr);
    tr(ow, owt, D, V, 1);

    k_embed<<<(MR * D + 255) / 256, 256, 0, stream>>>(x, tok, pos, h);

    auto gemm = [&](const __bf16* A, const __bf16* Bt, const float* bias, const float* res,
                    float* Cf, __bf16* Cb, int M, int N, int K, int relu) {
        dim3 grid(N / 64, M / 128);
        k_gemm_bf16<<<grid, 256, 0, stream>>>(A, Bt, bias, res, Cf, Cb, M, N, K, relu);
    };

    for (int l = 0; l < Lyr; l++) {
        k_layernorm_bf16<<<MR, 256, 0, stream>>>(h, g1 + l * D, b1 + l * D, hn);
        gemm(hn, qt + (size_t)l * D * D, nullptr, nullptr, nullptr, qbuf, MR, D, D, 0);
        gemm(hn, kt + (size_t)l * D * D, nullptr, nullptr, nullptr, kbuf, MR, D, D, 0);
        gemm(hn, vt + (size_t)l * D * D, nullptr, nullptr, nullptr, vbuf, MR, D, D, 0);
        k_attention<<<dim3(T / 16, Bsz * H), 32, 0, stream>>>(qbuf, kbuf, vbuf, attb);
        gemm(attb, pt + (size_t)l * D * D, pb + l * D, h, h, nullptr, MR, D, D, 0);
        k_layernorm_bf16<<<MR, 256, 0, stream>>>(h, g2 + l * D, b2 + l * D, hn);
        gemm(hn, w1t + (size_t)l * D * DFF, fb1 + l * DFF, nullptr, nullptr, f1b, MR, DFF, D, 1);
        gemm(f1b, w2t + (size_t)l * DFF * D, fb2 + l * D, h, h, nullptr, MR, D, DFF, 0);
    }

    k_layernorm_bf16<<<MR, 256, 0, stream>>>(h, gf, bf, hn);
    float* logits = (float*)d_out;
    gemm(hn, owt, ob, nullptr, logits, nullptr, MR, V, D, 0);

    float* loss = logits + (size_t)MR * V;
    hipMemsetAsync(loss, 0, sizeof(float), stream);
    k_loss<<<MR, 256, 0, stream>>>(logits, tgt, loss);
}
